// PointerGeneratorNetworks_16716012716063
// MI455X (gfx1250) — compile-verified, matrix operand order checked by reading, not run on hardware
//
#include <hip/hip_runtime.h>
#include <hip/hip_bf16.h>
#include <math.h>

// Problem sizes (match reference)
#define V_   50000
#define E_   128
#define H_   256
#define B_   16
#define S_   400
#define T_   64
#define NOOV_ 64
#define EPSL 1e-12f

typedef __attribute__((ext_vector_type(16))) _Float16 v16h;
typedef __attribute__((ext_vector_type(8)))  float    v8f;

// ---------------------------------------------------------------------------
// WMMA helpers (16x16x32 f16 -> f32), wave32 layouts per CDNA5 ISA 7.12.2
// A fragment: lane<16 -> row=lane, K chunks {k0..k0+7, k0+16..k0+23}
//             lane>=16 -> row=lane-16, K chunks {k0+8..k0+15, k0+24..k0+31}
// B loaded identically from W (N x K row-major) to compute A @ W^T.
// ---------------------------------------------------------------------------
__device__ __forceinline__ v16h ld_frag(const _Float16* base, int ld, int row0, int k0) {
    const int lane = threadIdx.x & 31;
    const _Float16* p = base + (size_t)(row0 + (lane & 15)) * ld + k0 + ((lane >> 4) << 3);
    v16h r;
#pragma unroll
    for (int i = 0; i < 8; ++i) r[i] = p[i];
#pragma unroll
    for (int i = 0; i < 8; ++i) r[8 + i] = p[16 + i];
    return r;
}

__device__ __forceinline__ v8f wmma32(v16h a, v16h b, v8f c) {
    return __builtin_amdgcn_wmma_f32_16x16x32_f16(false, a, false, b, (short)0, c, false, false);
}

// C/D layout: VGPR j -> (m = j + 8*(lane/16), n = lane%16)
__device__ __forceinline__ void st_tile_f32(float* dst, int ld, int row0, int col0, v8f c) {
    const int lane = threadIdx.x & 31;
    const int n  = col0 + (lane & 15);
    const int m0 = row0 + ((lane >> 4) << 3);
#pragma unroll
    for (int j = 0; j < 8; ++j) dst[(size_t)(m0 + j) * ld + n] = c[j];
}

// ---------------------------------------------------------------------------
// Block reductions (wave32; blockDim.x <= 512)
// ---------------------------------------------------------------------------
__device__ __forceinline__ float block_reduce_sum(float v, float* red) {
    __syncthreads();
    const int wave = threadIdx.x >> 5, lane = threadIdx.x & 31;
#pragma unroll
    for (int off = 16; off > 0; off >>= 1) v += __shfl_down(v, off);
    if (lane == 0) red[wave] = v;
    __syncthreads();
    float r = (threadIdx.x < (blockDim.x >> 5)) ? red[threadIdx.x] : 0.f;
    if (wave == 0) {
#pragma unroll
        for (int off = 16; off > 0; off >>= 1) r += __shfl_down(r, off);
        if (lane == 0) red[0] = r;
    }
    __syncthreads();
    return red[0];
}

__device__ __forceinline__ float block_reduce_max(float v, float* red) {
    __syncthreads();
    const int wave = threadIdx.x >> 5, lane = threadIdx.x & 31;
#pragma unroll
    for (int off = 16; off > 0; off >>= 1) v = fmaxf(v, __shfl_down(v, off));
    if (lane == 0) red[wave] = v;
    __syncthreads();
    float r = (threadIdx.x < (blockDim.x >> 5)) ? red[threadIdx.x] : -INFINITY;
    if (wave == 0) {
#pragma unroll
        for (int off = 16; off > 0; off >>= 1) r = fmaxf(r, __shfl_down(r, off));
        if (lane == 0) red[0] = r;
    }
    __syncthreads();
    return red[0];
}

__device__ __forceinline__ float sigm(float x) { return 1.f / (1.f + __expf(-x)); }

// ---------------------------------------------------------------------------
// Utility kernels
// ---------------------------------------------------------------------------
__global__ void k_f32_to_h(const float* __restrict__ src, _Float16* __restrict__ dst, int n) {
    int i = blockIdx.x * blockDim.x + threadIdx.x;
    if (i < n) dst[i] = (_Float16)src[i];
}

__global__ void k_copy_f32(const float* __restrict__ src, float* __restrict__ dst, int n) {
    int i = blockIdx.x * blockDim.x + threadIdx.x;
    if (i < n) dst[i] = src[i];
}

// Gather encoder embeddings -> (S, B, E) half
__global__ void k_embed_enc(const float* __restrict__ emb, const int* __restrict__ tok,
                            _Float16* __restrict__ xout) {
    int i = blockIdx.x * blockDim.x + threadIdx.x;
    if (i >= S_ * B_ * E_) return;
    int e = i % E_;
    int b = (i / E_) % B_;
    int s = i / (E_ * B_);
    int t = tok[b * S_ + s];
    xout[i] = (_Float16)emb[(size_t)t * E_ + e];
}

// ---------------------------------------------------------------------------
// Encoder LSTM: one block per direction, 16 waves. Wave w owns gate columns
// q in [16w, 16w+16) for all four gates (i/f/g/o) -> cell state c stays in
// VGPRs across all 400 steps; h is shared via LDS (half) for the K-loop.
// ---------------------------------------------------------------------------
__global__ __launch_bounds__(512)
void k_enc_lstm(const _Float16* __restrict__ xall,
                const _Float16* __restrict__ wih_f, const _Float16* __restrict__ whh_f,
                const float* __restrict__ b_f,
                const _Float16* __restrict__ wih_b, const _Float16* __restrict__ whh_b,
                const float* __restrict__ b_b,
                _Float16* __restrict__ enc_out_h,   // (B, S, 2H)
                float* __restrict__ h_stack,        // (2, B, H) flat: [hf, hb]
                float* __restrict__ c_stack)        // (2, B, H)
{
    const int dir = blockIdx.x;
    const _Float16* wih = dir ? wih_b : wih_f;
    const _Float16* whh = dir ? whh_b : whh_f;
    const float* bias   = dir ? b_b  : b_f;

    __shared__ _Float16 hsh[B_][H_ + 8];
    for (int i = threadIdx.x; i < B_ * (H_ + 8); i += blockDim.x)
        (&hsh[0][0])[i] = (_Float16)0.f;
    __syncthreads();

    const int wave = threadIdx.x >> 5;
    const int lane = threadIdx.x & 31;
    const int q  = wave * 16 + (lane & 15);  // column in [0,256)
    const int m0 = (lane >> 4) << 3;

    float creg[8];
#pragma unroll
    for (int j = 0; j < 8; ++j) creg[j] = 0.f;

    for (int step = 0; step < S_; ++step) {
        const int s = dir ? (S_ - 1 - step) : step;
        const _Float16* xt = xall + (size_t)s * B_ * E_;

        v8f acc[4];
#pragma unroll
        for (int g = 0; g < 4; ++g) {
            float bv = bias[g * H_ + q];
#pragma unroll
            for (int j = 0; j < 8; ++j) acc[g][j] = bv;
        }
        // x @ Wih^T  (K = 128)
#pragma unroll
        for (int kt = 0; kt < E_ / 32; ++kt) {
            v16h a = ld_frag(xt, E_, 0, kt * 32);
#pragma unroll
            for (int g = 0; g < 4; ++g) {
                v16h bf = ld_frag(wih, E_, g * H_ + wave * 16, kt * 32);
                acc[g] = wmma32(a, bf, acc[g]);
            }
        }
        // h @ Whh^T  (K = 256)
        for (int kt = 0; kt < H_ / 32; ++kt) {
            v16h a = ld_frag(&hsh[0][0], H_ + 8, 0, kt * 32);
#pragma unroll
            for (int g = 0; g < 4; ++g) {
                v16h bf = ld_frag(whh, H_, g * H_ + wave * 16, kt * 32);
                acc[g] = wmma32(a, bf, acc[g]);
            }
        }
        __syncthreads();  // all h reads complete before updating h
#pragma unroll
        for (int j = 0; j < 8; ++j) {
            const int m = m0 + j;
            float ig = sigm(acc[0][j]);
            float fg = sigm(acc[1][j]);
            float gg = tanhf(acc[2][j]);
            float og = sigm(acc[3][j]);
            float c  = fg * creg[j] + ig * gg;
            float h  = og * tanhf(c);
            creg[j]  = c;
            hsh[m][q] = (_Float16)h;
            enc_out_h[((size_t)m * S_ + s) * (2 * H_) + dir * H_ + q] = (_Float16)h;
        }
        __syncthreads();
    }
#pragma unroll
    for (int j = 0; j < 8; ++j) {
        const int m = m0 + j;
        h_stack[dir * B_ * H_ + m * H_ + q] = (float)hsh[m][q];
        c_stack[dir * B_ * H_ + m * H_ + q] = creg[j];
    }
}

// ---------------------------------------------------------------------------
// Reduce state: dec_h = relu(h_n @ red_h_w^T + b), dec_c likewise.
// h_n = stack([hf,hb]).reshape(B, 2H) == flat view of h_stack.
// ---------------------------------------------------------------------------
__global__ __launch_bounds__(512)
void k_reduce(const float* __restrict__ h_stack, const float* __restrict__ c_stack,
              const _Float16* __restrict__ red_h_w_h, const float* __restrict__ red_h_b,
              const _Float16* __restrict__ red_c_w_h, const float* __restrict__ red_c_b,
              _Float16* __restrict__ h_h_state, float* __restrict__ c_state,
              _Float16* __restrict__ hid_in_h)
{
    __shared__ _Float16 inh[B_][2 * H_];
    const int wave = threadIdx.x >> 5;
    const int lane = threadIdx.x & 31;
    for (int pass = 0; pass < 2; ++pass) {
        const float* src = pass ? c_stack : h_stack;
        for (int i = threadIdx.x; i < B_ * 2 * H_; i += blockDim.x)
            (&inh[0][0])[i] = (_Float16)src[i];
        __syncthreads();
        const _Float16* w = pass ? red_c_w_h : red_h_w_h;
        const float* bb   = pass ? red_c_b  : red_h_b;
        const int n = wave * 16 + (lane & 15);
        v8f acc;
        float bv = bb[n];
#pragma unroll
        for (int j = 0; j < 8; ++j) acc[j] = bv;
        for (int kt = 0; kt < (2 * H_) / 32; ++kt) {
            v16h a  = ld_frag(&inh[0][0], 2 * H_, 0, kt * 32);
            v16h bf = ld_frag(w, 2 * H_, wave * 16, kt * 32);
            acc = wmma32(a, bf, acc);
        }
        const int m0 = (lane >> 4) << 3;
#pragma unroll
        for (int j = 0; j < 8; ++j) {
            float v = fmaxf(acc[j], 0.f);
            int m = m0 + j;
            if (pass == 0) {
                h_h_state[m * H_ + n] = (_Float16)v;
                hid_in_h[m * 3 * H_ + n] = (_Float16)v;
            } else {
                c_state[m * H_ + n] = v;
            }
        }
        __syncthreads();
    }
}

// ---------------------------------------------------------------------------
// enc_feat = enc_out @ att_wh^T : GEMM M = B*S = 6400, N = 512, K = 512.
// 8 waves/block, one 16x16 tile per wave. 1600 blocks.
// ---------------------------------------------------------------------------
__global__ __launch_bounds__(256)
void k_encfeat(const _Float16* __restrict__ enc_out_h,
               const _Float16* __restrict__ att_wh_h,
               float* __restrict__ enc_feat)
{
    const int tile = blockIdx.x * 8 + (threadIdx.x >> 5);
    const int mt = tile >> 5;   // 0..399
    const int nt = tile & 31;   // 0..31
    v8f acc;
#pragma unroll
    for (int j = 0; j < 8; ++j) acc[j] = 0.f;
    const _Float16* A = enc_out_h + (size_t)mt * 16 * (2 * H_);
    for (int kt = 0; kt < 16; ++kt) {
        v16h a  = ld_frag(A, 2 * H_, 0, kt * 32);
        v16h bf = ld_frag(att_wh_h, 2 * H_, nt * 16, kt * 32);
        acc = wmma32(a, bf, acc);
    }
    st_tile_f32(enc_feat + (size_t)mt * 16 * (2 * H_), 2 * H_, 0, nt * 16, acc);
}

// ---------------------------------------------------------------------------
// Decoder cell (one block, 16 waves): x = [ctx, emb] @ lin_in^T + b;
// LSTM cell (c in ws, h half in ws); status = [h,c]; sproj = status@att_ws^T;
// gp_partial = gp_ws . status + gp_wx . x + gp_b.
// ---------------------------------------------------------------------------
__global__ __launch_bounds__(512)
void k_dec_cell(int t,
                const float* __restrict__ ctx_f,
                const int* __restrict__ dec_input,
                const float* __restrict__ dec_embed,
                const _Float16* __restrict__ lin_in_w_h, const float* __restrict__ lin_in_b,
                const _Float16* __restrict__ dec_wih_h, const _Float16* __restrict__ dec_whh_h,
                const float* __restrict__ dec_b,
                const _Float16* __restrict__ att_ws_h,
                const float* __restrict__ gp_ws, const float* __restrict__ gp_wx,
                const float* __restrict__ gp_b,
                _Float16* __restrict__ h_h_state, float* __restrict__ c_state,
                _Float16* __restrict__ hid_in_h,
                float* __restrict__ sproj_f, float* __restrict__ gp_partial)
{
    __shared__ _Float16 xin[B_][2 * H_ + E_];  // 16 x 640
    __shared__ _Float16 xh[B_][E_];
    __shared__ float    xf[B_][E_];
    __shared__ _Float16 st[B_][2 * H_];        // status [h, c] half
    const int wave = threadIdx.x >> 5;
    const int lane = threadIdx.x & 31;

    // build [ctx, emb]
    for (int i = threadIdx.x; i < B_ * (2 * H_ + E_); i += blockDim.x) {
        int b = i / (2 * H_ + E_), k = i % (2 * H_ + E_);
        float v;
        if (k < 2 * H_) v = ctx_f[b * 2 * H_ + k];
        else {
            int tok = dec_input[b * T_ + t];
            v = dec_embed[(size_t)tok * E_ + (k - 2 * H_)];
        }
        xin[b][k] = (_Float16)v;
    }
    __syncthreads();

    // x = xin @ lin_in_w^T + b   (N = 128 -> 8 tiles on waves 0..7, K = 640)
    if (wave < 8) {
        const int n = wave * 16 + (lane & 15);
        v8f acc;
        float bv = lin_in_b[n];
#pragma unroll
        for (int j = 0; j < 8; ++j) acc[j] = bv;
        for (int kt = 0; kt < (2 * H_ + E_) / 32; ++kt) {
            v16h a  = ld_frag(&xin[0][0], 2 * H_ + E_, 0, kt * 32);
            v16h bf = ld_frag(lin_in_w_h, 2 * H_ + E_, wave * 16, kt * 32);
            acc = wmma32(a, bf, acc);
        }
        const int m0 = (lane >> 4) << 3;
#pragma unroll
        for (int j = 0; j < 8; ++j) {
            int m = m0 + j;
            xh[m][n] = (_Float16)acc[j];
            xf[m][n] = acc[j];
        }
    }
    __syncthreads();

    // gates: wave w owns columns [16w,16w+16) of every gate
    const int q  = wave * 16 + (lane & 15);
    const int m0 = (lane >> 4) << 3;
    v8f acc[4];
#pragma unroll
    for (int g = 0; g < 4; ++g) {
        float bv = dec_b[g * H_ + q];
#pragma unroll
        for (int j = 0; j < 8; ++j) acc[g][j] = bv;
    }
#pragma unroll
    for (int kt = 0; kt < E_ / 32; ++kt) {
        v16h a = ld_frag(&xh[0][0], E_, 0, kt * 32);
#pragma unroll
        for (int g = 0; g < 4; ++g) {
            v16h bf = ld_frag(dec_wih_h, E_, g * H_ + wave * 16, kt * 32);
            acc[g] = wmma32(a, bf, acc[g]);
        }
    }
    for (int kt = 0; kt < H_ / 32; ++kt) {
        v16h a = ld_frag(h_h_state, H_, 0, kt * 32);
#pragma unroll
        for (int g = 0; g < 4; ++g) {
            v16h bf = ld_frag(dec_whh_h, H_, g * H_ + wave * 16, kt * 32);
            acc[g] = wmma32(a, bf, acc[g]);
        }
    }
    __syncthreads();  // all reads of old h done before rewrite
#pragma unroll
    for (int j = 0; j < 8; ++j) {
        const int m = m0 + j;
        float ig = sigm(acc[0][j]);
        float fg = sigm(acc[1][j]);
        float gg = tanhf(acc[2][j]);
        float og = sigm(acc[3][j]);
        float c  = fg * c_state[m * H_ + q] + ig * gg;
        float h  = og * tanhf(c);
        c_state[m * H_ + q] = c;
        _Float16 hh = (_Float16)h;
        st[m][q]       = hh;
        st[m][H_ + q]  = (_Float16)c;
        h_h_state[m * H_ + q]   = hh;
        hid_in_h[m * 3 * H_ + q] = hh;
    }
    __syncthreads();

    // sproj = status @ att_ws^T  (N = 512 -> 2 tiles per wave, K = 512)
    for (int rep = 0; rep < 2; ++rep) {
        const int nt = wave * 2 + rep;
        v8f a2;
#pragma unroll
        for (int j = 0; j < 8; ++j) a2[j] = 0.f;
        for (int kt = 0; kt < 16; ++kt) {
            v16h a  = ld_frag(&st[0][0], 2 * H_, 0, kt * 32);
            v16h bf = ld_frag(att_ws_h, 2 * H_, nt * 16, kt * 32);
            a2 = wmma32(a, bf, a2);
        }
        st_tile_f32(sproj_f, 2 * H_, 0, nt * 16, a2);
    }

    // gp_partial: wave b handles batch b
    {
        const int b = wave;
        float s = 0.f;
        for (int k = lane; k < 2 * H_; k += 32) s += gp_ws[k] * (float)st[b][k];
        for (int k = lane; k < E_;    k += 32) s += gp_wx[k] * xf[b][k];
#pragma unroll
        for (int off = 16; off > 0; off >>= 1) s += __shfl_down(s, off);
        if (lane == 0) gp_partial[b] = s + gp_b[0];
    }
}

// ---------------------------------------------------------------------------
// Attention scores, softmax over S, context vector, gen_p. One block per b.
// ---------------------------------------------------------------------------
__global__ __launch_bounds__(512)
void k_attn(const float* __restrict__ enc_feat, const float* __restrict__ sproj_f,
            const float* __restrict__ att_v, const float* __restrict__ enc_mask,
            const _Float16* __restrict__ enc_out_h,
            const float* __restrict__ gp_wh, const float* __restrict__ gp_partial,
            float* __restrict__ ctx_f, _Float16* __restrict__ hid_in_h,
            float* __restrict__ attn_w, float* __restrict__ gen_p)
{
    const int b = blockIdx.x;
    __shared__ float e[S_];
    __shared__ float red[16];
    __shared__ float ctxs[2 * H_];
    const int wave = threadIdx.x >> 5;
    const int lane = threadIdx.x & 31;

    for (int s = wave; s < S_; s += 16) {
        const float* ef = enc_feat + ((size_t)b * S_ + s) * 2 * H_;
        float sum = 0.f;
        for (int k = lane; k < 2 * H_; k += 32)
            sum += att_v[k] * tanhf(ef[k] + sproj_f[b * 2 * H_ + k]);
#pragma unroll
        for (int off = 16; off > 0; off >>= 1) sum += __shfl_down(sum, off);
        if (lane == 0)
            e[s] = (enc_mask[b * S_ + s] == 0.f) ? -INFINITY : sum;
    }
    __syncthreads();

    float lm = -INFINITY;
    for (int s = threadIdx.x; s < S_; s += blockDim.x) lm = fmaxf(lm, e[s]);
    float mx = block_reduce_max(lm, red);
    float ls = 0.f;
    for (int s = threadIdx.x; s < S_; s += blockDim.x) ls += __expf(e[s] - mx);
    float tot = block_reduce_sum(ls, red);
    for (int s = threadIdx.x; s < S_; s += blockDim.x) e[s] = __expf(e[s] - mx) / tot;
    __syncthreads();
    for (int s = threadIdx.x; s < S_; s += blockDim.x) attn_w[b * S_ + s] = e[s];

    // ctx[b, d] = sum_s a[s] * enc_out[b, s, d]
    for (int d = threadIdx.x; d < 2 * H_; d += blockDim.x) {
        float sum = 0.f;
        for (int s = 0; s < S_; ++s)
            sum += e[s] * (float)enc_out_h[((size_t)b * S_ + s) * 2 * H_ + d];
        ctxs[d] = sum;
        ctx_f[b * 2 * H_ + d] = sum;
        hid_in_h[b * 3 * H_ + H_ + d] = (_Float16)sum;
    }
    __syncthreads();

    float g = 0.f;
    for (int d = threadIdx.x; d < 2 * H_; d += blockDim.x) g += ctxs[d] * gp_wh[d];
    float gt = block_reduce_sum(g, red);
    if (threadIdx.x == 0) gen_p[b] = sigm(gp_partial[b] + gt);
}

// hid = relu([h, ctx] @ out_w1^T + b1) : M=16, N=256, K=768
__global__ __launch_bounds__(512)
void k_out_hidden(const _Float16* __restrict__ hid_in_h,
                  const _Float16* __restrict__ out_w1_h, const float* __restrict__ out_b1,
                  _Float16* __restrict__ hid_h)
{
    const int wave = threadIdx.x >> 5;
    const int lane = threadIdx.x & 31;
    const int n = wave * 16 + (lane & 15);
    v8f acc;
    float bv = out_b1[n];
#pragma unroll
    for (int j = 0; j < 8; ++j) acc[j] = bv;
    for (int kt = 0; kt < (3 * H_) / 32; ++kt) {
        v16h a  = ld_frag(hid_in_h, 3 * H_, 0, kt * 32);
        v16h bf = ld_frag(out_w1_h, 3 * H_, wave * 16, kt * 32);
        acc = wmma32(a, bf, acc);
    }
    const int m0 = (lane >> 4) << 3;
#pragma unroll
    for (int j = 0; j < 8; ++j)
        hid_h[(m0 + j) * H_ + n] = (_Float16)fmaxf(acc[j], 0.f);
}

// logits = hid @ out_w2^T + b2 : M=16, N=50000 (3125 tiles), K=256
__global__ __launch_bounds__(256)
void k_logits(const _Float16* __restrict__ hid_h, const _Float16* __restrict__ out_w2_h,
              const float* __restrict__ out_b2, float* __restrict__ logits)
{
    const int tile = blockIdx.x * 8 + (threadIdx.x >> 5);
    if (tile >= V_ / 16) return;
    const int lane = threadIdx.x & 31;
    const int n = tile * 16 + (lane & 15);
    v8f acc;
    float bv = out_b2[n];
#pragma unroll
    for (int j = 0; j < 8; ++j) acc[j] = bv;
    if (tile + 1 < V_ / 16)
        __builtin_prefetch(out_w2_h + (size_t)(tile + 1) * 16 * H_, 0, 1);
    for (int kt = 0; kt < H_ / 32; ++kt) {
        v16h a  = ld_frag(hid_h, H_, 0, kt * 32);
        v16h bf = ld_frag(out_w2_h, H_, tile * 16, kt * 32);
        acc = wmma32(a, bf, acc);
    }
    st_tile_f32(logits, V_, 0, tile * 16, acc);
}

// Final dist is only read at the target index, so compute softmax(tgt) +
// masked attention sum directly: no (B, V+NOOV) scatter needed.
__global__ __launch_bounds__(512)
void k_loss_step(int t, const float* __restrict__ logits, const float* __restrict__ attn_w,
                 const float* __restrict__ gen_p, const int* __restrict__ enc_with_oov,
                 const int* __restrict__ dec_target, float* __restrict__ step_loss)
{
    const int b = blockIdx.x;
    __shared__ float red[16];
    const float* lg = logits + (size_t)b * V_;
    float lm = -INFINITY;
    for (int i = threadIdx.x; i < V_; i += blockDim.x) lm = fmaxf(lm, lg[i]);
    float mx = block_reduce_max(lm, red);
    float ls = 0.f;
    for (int i = threadIdx.x; i < V_; i += blockDim.x) ls += __expf(lg[i] - mx);
    float tot = block_reduce_sum(ls, red);
    const int tgt = dec_target[b * T_ + t];
    float cp = 0.f;
    for (int s = threadIdx.x; s < S_; s += blockDim.x)
        if (enc_with_oov[b * S_ + s] == tgt) cp += attn_w[b * S_ + s];
    cp = block_reduce_sum(cp, red);
    if (threadIdx.x == 0) {
        float gp = gen_p[b];
        float vp = (tgt < V_) ? gp * __expf(lg[tgt] - mx) / tot : 0.f;
        step_loss[t * B_ + b] = -logf(vp + (1.f - gp) * cp + EPSL);
    }
}

__global__ void k_final(const float* __restrict__ step_loss, const float* __restrict__ dec_mask,
                        float* __restrict__ out)
{
    __shared__ float acc[B_];
    const int b = threadIdx.x;
    if (b < B_) {
        float s = 0.f, ms = 0.f;
        for (int t = 0; t < T_; ++t) {
            float m = dec_mask[b * T_ + t];
            s  += step_loss[t * B_ + b] * m;
            ms += m;
        }
        acc[b] = s / ms;
    }
    __syncthreads();
    if (threadIdx.x == 0) {
        float tot = 0.f;
        for (int i = 0; i < B_; ++i) tot += acc[i];
        out[0] = tot / B_;
    }
}

// ---------------------------------------------------------------------------
// Host orchestration
// ---------------------------------------------------------------------------
extern "C" void kernel_launch(void* const* d_in, const int* in_sizes, int n_in,
                              void* d_out, int out_size, void* d_ws, size_t ws_size,
                              hipStream_t stream) {
    // params (flattened dict order), then remaining inputs
    const float* enc_embed = (const float*)d_in[0];
    const float* enc_wih_f = (const float*)d_in[1];
    const float* enc_whh_f = (const float*)d_in[2];
    const float* enc_b_f   = (const float*)d_in[3];
    const float* enc_wih_b = (const float*)d_in[4];
    const float* enc_whh_b = (const float*)d_in[5];
    const float* enc_b_b   = (const float*)d_in[6];
    const float* red_h_w   = (const float*)d_in[7];
    const float* red_h_b   = (const float*)d_in[8];
    const float* red_c_w   = (const float*)d_in[9];
    const float* red_c_b   = (const float*)d_in[10];
    const float* dec_embed = (const float*)d_in[11];
    const float* lin_in_w  = (const float*)d_in[12];
    const float* lin_in_b  = (const float*)d_in[13];
    const float* dec_wih   = (const float*)d_in[14];
    const float* dec_whh   = (const float*)d_in[15];
    const float* dec_b     = (const float*)d_in[16];
    const float* att_wh    = (const float*)d_in[17];
    const float* att_ws    = (const float*)d_in[18];
    const float* att_v     = (const float*)d_in[19];
    const float* gp_wh     = (const float*)d_in[20];
    const float* gp_ws     = (const float*)d_in[21];
    const float* gp_wx     = (const float*)d_in[22];
    const float* gp_b      = (const float*)d_in[23];
    const float* out_w1    = (const float*)d_in[24];
    const float* out_b1    = (const float*)d_in[25];
    const float* out_w2    = (const float*)d_in[26];
    const float* out_b2    = (const float*)d_in[27];
    const int*   encoder_input    = (const int*)d_in[28];
    const float* encoder_mask     = (const float*)d_in[29];
    const int*   encoder_with_oov = (const int*)d_in[30];
    const float* context_vec      = (const float*)d_in[32];
    const int*   decoder_input    = (const int*)d_in[34];
    const float* decoder_mask     = (const float*)d_in[35];
    const int*   decoder_target   = (const int*)d_in[36];

    // workspace bump allocator
    size_t off = 0;
    auto alloc = [&](size_t bytes) -> void* {
        void* p = (char*)d_ws + off;
        off = (off + bytes + 255) & ~(size_t)255;
        return p;
    };
    _Float16* wih_f_h    = (_Float16*)alloc((size_t)4 * H_ * E_ * 2);
    _Float16* whh_f_h    = (_Float16*)alloc((size_t)4 * H_ * H_ * 2);
    _Float16* wih_b_h    = (_Float16*)alloc((size_t)4 * H_ * E_ * 2);
    _Float16* whh_b_h    = (_Float16*)alloc((size_t)4 * H_ * H_ * 2);
    _Float16* red_h_w_h  = (_Float16*)alloc((size_t)H_ * 2 * H_ * 2);
    _Float16* red_c_w_h  = (_Float16*)alloc((size_t)H_ * 2 * H_ * 2);
    _Float16* lin_in_w_h = (_Float16*)alloc((size_t)E_ * (2 * H_ + E_) * 2);
    _Float16* dec_wih_h  = (_Float16*)alloc((size_t)4 * H_ * E_ * 2);
    _Float16* dec_whh_h  = (_Float16*)alloc((size_t)4 * H_ * H_ * 2);
    _Float16* att_wh_h   = (_Float16*)alloc((size_t)2 * H_ * 2 * H_ * 2);
    _Float16* att_ws_h   = (_Float16*)alloc((size_t)2 * H_ * 2 * H_ * 2);
    _Float16* out_w1_h   = (_Float16*)alloc((size_t)H_ * 3 * H_ * 2);
    _Float16* out_w2_h   = (_Float16*)alloc((size_t)V_ * H_ * 2);
    _Float16* x_enc_h    = (_Float16*)alloc((size_t)S_ * B_ * E_ * 2);
    _Float16* enc_out_h  = (_Float16*)alloc((size_t)B_ * S_ * 2 * H_ * 2);
    float*    enc_feat   = (float*)alloc((size_t)B_ * S_ * 2 * H_ * 4);
    float*    h_stack    = (float*)alloc((size_t)2 * B_ * H_ * 4);
    float*    c_stack    = (float*)alloc((size_t)2 * B_ * H_ * 4);
    _Float16* h_h_state  = (_Float16*)alloc((size_t)B_ * H_ * 2);
    float*    c_state    = (float*)alloc((size_t)B_ * H_ * 4);
    float*    ctx_f      = (float*)alloc((size_t)B_ * 2 * H_ * 4);
    _Float16* hid_in_h   = (_Float16*)alloc((size_t)B_ * 3 * H_ * 2);
    float*    sproj_f    = (float*)alloc((size_t)B_ * 2 * H_ * 4);
    float*    gp_partial = (float*)alloc((size_t)B_ * 4);
    float*    attn_w     = (float*)alloc((size_t)B_ * S_ * 4);
    float*    gen_p      = (float*)alloc((size_t)B_ * 4);
    _Float16* hid_h      = (_Float16*)alloc((size_t)B_ * H_ * 2);
    float*    logits     = (float*)alloc((size_t)B_ * V_ * 4);
    float*    step_loss  = (float*)alloc((size_t)T_ * B_ * 4);
    (void)ws_size; (void)n_in; (void)in_sizes; (void)out_size;

    auto conv = [&](const float* s, _Float16* d, int n) {
        k_f32_to_h<<<(n + 255) / 256, 256, 0, stream>>>(s, d, n);
    };
    conv(enc_wih_f, wih_f_h,   4 * H_ * E_);
    conv(enc_whh_f, whh_f_h,   4 * H_ * H_);
    conv(enc_wih_b, wih_b_h,   4 * H_ * E_);
    conv(enc_whh_b, whh_b_h,   4 * H_ * H_);
    conv(red_h_w,   red_h_w_h, H_ * 2 * H_);
    conv(red_c_w,   red_c_w_h, H_ * 2 * H_);
    conv(lin_in_w,  lin_in_w_h, E_ * (2 * H_ + E_));
    conv(dec_wih,   dec_wih_h, 4 * H_ * E_);
    conv(dec_whh,   dec_whh_h, 4 * H_ * H_);
    conv(att_wh,    att_wh_h,  2 * H_ * 2 * H_);
    conv(att_ws,    att_ws_h,  2 * H_ * 2 * H_);
    conv(out_w1,    out_w1_h,  H_ * 3 * H_);
    conv(out_w2,    out_w2_h,  V_ * H_);

    // encoder embeddings gather -> (S,B,E) half
    {
        int n = S_ * B_ * E_;
        k_embed_enc<<<(n + 255) / 256, 256, 0, stream>>>(enc_embed, encoder_input, x_enc_h);
    }

    // bidirectional encoder LSTM (block 0 fwd, block 1 bwd)
    k_enc_lstm<<<2, 512, 0, stream>>>(x_enc_h, wih_f_h, whh_f_h, enc_b_f,
                                      wih_b_h, whh_b_h, enc_b_b,
                                      enc_out_h, h_stack, c_stack);

    // reduce state -> decoder initial h, c
    k_reduce<<<1, 512, 0, stream>>>(h_stack, c_stack, red_h_w_h, red_h_b,
                                    red_c_w_h, red_c_b, h_h_state, c_state, hid_in_h);

    // enc_feat GEMM (step-invariant attention features)
    k_encfeat<<<(B_ * S_ / 16) * (2 * H_ / 16) / 8, 256, 0, stream>>>(enc_out_h, att_wh_h, enc_feat);

    // initial context vector
    k_copy_f32<<<(B_ * 2 * H_ + 255) / 256, 256, 0, stream>>>(context_vec, ctx_f, B_ * 2 * H_);

    // decoder steps
    for (int t = 0; t < T_; ++t) {
        k_dec_cell<<<1, 512, 0, stream>>>(t, ctx_f, decoder_input, dec_embed,
                                          lin_in_w_h, lin_in_b, dec_wih_h, dec_whh_h, dec_b,
                                          att_ws_h, gp_ws, gp_wx, gp_b,
                                          h_h_state, c_state, hid_in_h, sproj_f, gp_partial);
        k_attn<<<B_, 512, 0, stream>>>(enc_feat, sproj_f, att_v, encoder_mask, enc_out_h,
                                       gp_wh, gp_partial, ctx_f, hid_in_h, attn_w, gen_p);
        k_out_hidden<<<1, 512, 0, stream>>>(hid_in_h, out_w1_h, out_b1, hid_h);
        k_logits<<<(V_ / 16 + 7) / 8, 256, 0, stream>>>(hid_h, out_w2_h, out_b2, logits);
        k_loss_step<<<B_, 512, 0, stream>>>(t, logits, attn_w, gen_p,
                                            encoder_with_oov, decoder_target, step_loss);
    }

    k_final<<<1, 32, 0, stream>>>(step_loss, decoder_mask, (float*)d_out);
}
